// MacridVAE_35570919145766
// MI455X (gfx1250) — matrix operand order repaired and checked by reading, not compiled, not run on hardware
//
#include <hip/hip_runtime.h>
#include <math.h>

typedef __bf16 bf16_t;
typedef __attribute__((ext_vector_type(16))) __bf16 v16bf;
typedef __attribute__((ext_vector_type(8)))  __bf16 v8bf;
typedef __attribute__((ext_vector_type(8)))  float  v8f;

// ---------- helpers ----------
__device__ __forceinline__ bf16_t f2bf(float f) {
    union { float f; unsigned u; } in; in.f = f;
    unsigned u = in.u;
    unsigned r = (u + 0x7FFFu + ((u >> 16) & 1u)) >> 16;   // round to nearest even
    union { unsigned short s; bf16_t b; } out; out.s = (unsigned short)r;
    return out.b;
}

#define DCONST 64
#define KCONST 4

// ---------- K0: normalize prototypes ----------
__global__ void proto_norm_kernel(float* __restrict__ protoN, const float* __restrict__ proto) {
    int k = threadIdx.x;
    if (k >= KCONST) return;
    float s = 0.f;
    #pragma unroll
    for (int d = 0; d < DCONST; ++d) { float v = proto[k * DCONST + d]; s += v * v; }
    float inv = 1.f / fmaxf(sqrtf(s), 1e-12f);
    #pragma unroll
    for (int d = 0; d < DCONST; ++d) protoN[k * DCONST + d] = proto[k * DCONST + d] * inv;
}

// ---------- K1: normalize items + softmax cates ----------
__global__ __launch_bounds__(256) void items_cates_kernel(
    float* __restrict__ items, float* __restrict__ cates,
    const float* __restrict__ item_emb, const float* __restrict__ protoN, int N) {
    int n = blockIdx.x * blockDim.x + threadIdx.x;
    if (n >= N) return;
    float v[DCONST];
    float s = 0.f;
    #pragma unroll
    for (int d = 0; d < DCONST; ++d) { v[d] = item_emb[(size_t)n * DCONST + d]; s += v[d] * v[d]; }
    float inv = 1.f / fmaxf(sqrtf(s), 1e-12f);
    #pragma unroll
    for (int d = 0; d < DCONST; ++d) { v[d] *= inv; items[(size_t)n * DCONST + d] = v[d]; }
    float lg[KCONST]; float mx = -1e30f;
    #pragma unroll
    for (int k = 0; k < KCONST; ++k) {
        float dot = 0.f;
        #pragma unroll
        for (int d = 0; d < DCONST; ++d) dot += v[d] * protoN[k * DCONST + d];
        lg[k] = dot; mx = fmaxf(mx, dot);
    }
    float se = 0.f;
    #pragma unroll
    for (int k = 0; k < KCONST; ++k) { lg[k] = expf(lg[k] - mx); se += lg[k]; }
    float rse = 1.f / se;
    #pragma unroll
    for (int k = 0; k < KCONST; ++k) cates[(size_t)n * KCONST + k] = lg[k] * rse;
}

// ---------- K2: transpose f32 -> bf16 (for adjT) ----------
__global__ __launch_bounds__(256) void transpose_to_bf16(
    bf16_t* __restrict__ out, const float* __restrict__ in, int N) {
    __shared__ float tile[32][33];
    int bx = blockIdx.x * 32, by = blockIdx.y * 32;
    int tx = threadIdx.x & 31, ty = threadIdx.x >> 5;   // 32 x 8
    #pragma unroll
    for (int i = 0; i < 32; i += 8)
        tile[ty + i][tx] = in[(size_t)(by + ty + i) * N + bx + tx];
    __syncthreads();
    #pragma unroll
    for (int i = 0; i < 32; i += 8)
        out[(size_t)(bx + ty + i) * N + by + tx] = f2bf(tile[tx][ty + i]);
}

// ---------- K3: row norms of x_k ----------
__global__ __launch_bounds__(256) void xk_norm_kernel(
    float* __restrict__ invnorm, const float* __restrict__ rating,
    const float* __restrict__ cates, int B, int N) {
    int kb = blockIdx.x; int k = kb / B; int b = kb % B;
    const float* r = rating + (size_t)b * N;
    float s = 0.f;
    for (int n = threadIdx.x; n < N; n += 256) {
        float x = r[n] * cates[(size_t)n * KCONST + k];
        s += x * x;
    }
    __shared__ float red[256];
    red[threadIdx.x] = s; __syncthreads();
    for (int off = 128; off > 0; off >>= 1) {
        if (threadIdx.x < off) red[threadIdx.x] += red[threadIdx.x + off];
        __syncthreads();
    }
    if (threadIdx.x == 0) invnorm[kb] = 1.f / fmaxf(sqrtf(red[0]), 1e-12f);
}

// ---------- K4: build normalized x_k in bf16 ----------
__global__ __launch_bounds__(256) void xk_build_kernel(
    bf16_t* __restrict__ xk, const float* __restrict__ rating,
    const float* __restrict__ cates, const float* __restrict__ invnorm,
    int B, int N) {
    size_t idx = (size_t)blockIdx.x * blockDim.x + threadIdx.x;
    size_t total = (size_t)KCONST * B * N;
    if (idx >= total) return;
    int n = (int)(idx % N); size_t t = idx / N;
    int b = (int)(t % B); int k = (int)(t / B);
    float v = rating[(size_t)b * N + n] * cates[(size_t)n * KCONST + k] * invnorm[k * B + b];
    xk[idx] = f2bf(v);
}

// ---------- K5: generic f32 -> bf16 ----------
__global__ __launch_bounds__(256) void f2bf_kernel(
    bf16_t* __restrict__ out, const float* __restrict__ in, size_t total) {
    size_t idx = (size_t)blockIdx.x * blockDim.x + threadIdx.x;
    if (idx >= total) return;
    out[idx] = f2bf(in[idx]);
}

// ---------- K6: x = (tmp2 + x_k)/2, to bf16 ----------
__global__ __launch_bounds__(256) void combine_kernel(
    bf16_t* __restrict__ xbf, const float* __restrict__ tmp2,
    const float* __restrict__ rating, const float* __restrict__ cates,
    const float* __restrict__ invnorm, int B, int N) {
    size_t idx = (size_t)blockIdx.x * blockDim.x + threadIdx.x;
    size_t total = (size_t)KCONST * B * N;
    if (idx >= total) return;
    int n = (int)(idx % N); size_t t = idx / N;
    int b = (int)(t % B); int k = (int)(t / B);
    float xv = rating[(size_t)b * N + n] * cates[(size_t)n * KCONST + k] * invnorm[k * B + b];
    xbf[idx] = f2bf(0.5f * (tmp2[idx] + xv));
}

// ---------- WMMA bf16 GEMM: C[M,N] = A[M,K] * BT[N,K]^T ----------
// Block tile 128x128, BK=32, 256 threads = 8 waves (wave32), waves 2x4,
// each wave owns 64x32 -> 8x v_wmma_f32_16x16x32_bf16 per k-step.
// Tile staging uses CDNA5 GLOBAL_LOAD_ASYNC_TO_LDS_B128 (ASYNCcnt-tracked DMA,
// no VGPR staging), double-buffered, drained with s_wait_asynccnt before barrier.
#define BM 128
#define BN 128
#define BK 32

__global__ __launch_bounds__(256) void gemm_bf16_wmma(
    float* __restrict__ Cf, bf16_t* __restrict__ Cb,
    const bf16_t* __restrict__ A, const bf16_t* __restrict__ BT,
    int M, int N, int K) {
    __shared__ __align__(16) bf16_t As[2][BM * BK];
    __shared__ __align__(16) bf16_t Bs[2][BM * BK];

    const int tid  = threadIdx.x;
    const int wave = tid >> 5;
    const int lane = tid & 31;
    const int wm   = wave >> 2;          // 0..1  (64 rows each)
    const int wn   = wave & 3;           // 0..3  (32 cols each)
    const int hi   = lane >> 4;          // lane group
    const int l16  = lane & 15;

    const int rowBlock = blockIdx.y * BM;
    const int colBlock = blockIdx.x * BN;

    v8f acc[4][2];
    #pragma unroll
    for (int mt = 0; mt < 4; ++mt)
        #pragma unroll
        for (int nt = 0; nt < 2; ++nt)
            acc[mt][nt] = (v8f){0.f,0.f,0.f,0.f,0.f,0.f,0.f,0.f};

    // staging: thread handles rows r0 and r0+64, 8-bf16 chunk q (16B each)
    const int r0 = tid >> 2;
    const int q8 = (tid & 3) * 8;
    const bf16_t* aG = A  + (size_t)(rowBlock + r0) * K + q8;
    const bf16_t* bG = BT + (size_t)(colBlock + r0) * K + q8;
    const size_t stride64 = (size_t)64 * K;

    // per-thread LDS destination byte offsets (low 32 bits of LDS-aperture address)
    const unsigned dA0 = (unsigned)(uintptr_t)(void*)&As[0][r0 * BK + q8];
    const unsigned dA1 = (unsigned)(uintptr_t)(void*)&As[0][(r0 + 64) * BK + q8];
    const unsigned dB0 = (unsigned)(uintptr_t)(void*)&Bs[0][r0 * BK + q8];
    const unsigned dB1 = (unsigned)(uintptr_t)(void*)&Bs[0][(r0 + 64) * BK + q8];
    const unsigned bufBytes = (unsigned)(BM * BK * sizeof(bf16_t));   // 8192

    const int steps = K / BK;
    int buf = 0;

    // prologue: DMA tile 0 into buffer 0
    {
        asm volatile("global_load_async_to_lds_b128 %0, %1, off"
                     :: "v"(dA0), "v"(aG) : "memory");
        asm volatile("global_load_async_to_lds_b128 %0, %1, off"
                     :: "v"(dA1), "v"(aG + stride64) : "memory");
        asm volatile("global_load_async_to_lds_b128 %0, %1, off"
                     :: "v"(dB0), "v"(bG) : "memory");
        asm volatile("global_load_async_to_lds_b128 %0, %1, off"
                     :: "v"(dB1), "v"(bG + stride64) : "memory");
        asm volatile("s_wait_asynccnt 0x0" ::: "memory");
    }
    __syncthreads();

    for (int kt = 0; kt < steps; ++kt) {
        const bool hasNext = (kt + 1) < steps;
        if (hasNext) {
            const size_t   ko  = (size_t)(kt + 1) * BK;
            const unsigned off = (unsigned)((buf ^ 1) ? bufBytes : 0u);
            asm volatile("global_load_async_to_lds_b128 %0, %1, off"
                         :: "v"(dA0 + off), "v"(aG + ko) : "memory");
            asm volatile("global_load_async_to_lds_b128 %0, %1, off"
                         :: "v"(dA1 + off), "v"(aG + stride64 + ko) : "memory");
            asm volatile("global_load_async_to_lds_b128 %0, %1, off"
                         :: "v"(dB0 + off), "v"(bG + ko) : "memory");
            asm volatile("global_load_async_to_lds_b128 %0, %1, off"
                         :: "v"(dB1 + off), "v"(bG + stride64 + ko) : "memory");
        }

        // fragment loads from LDS (current buffer)
        v16bf af[4], bfm[2];
        #pragma unroll
        for (int mt = 0; mt < 4; ++mt) {
            const bf16_t* p = &As[buf][(wm * 64 + mt * 16 + l16) * BK + hi * 8];
            v8bf lo = *(const v8bf*)(p);        // K 0..7 / 8..15
            v8bf hh = *(const v8bf*)(p + 16);   // K 16..23 / 24..31
            af[mt] = __builtin_shufflevector(lo, hh, 0,1,2,3,4,5,6,7,8,9,10,11,12,13,14,15);
        }
        #pragma unroll
        for (int nt = 0; nt < 2; ++nt) {
            const bf16_t* p = &Bs[buf][(wn * 32 + nt * 16 + l16) * BK + hi * 8];
            v8bf lo = *(const v8bf*)(p);        // K 0..7 / 8..15
            v8bf hh = *(const v8bf*)(p + 16);   // K 16..23 / 24..31
            bfm[nt] = __builtin_shufflevector(lo, hh, 0,1,2,3,4,5,6,7,8,9,10,11,12,13,14,15);
        }

        #pragma unroll
        for (int mt = 0; mt < 4; ++mt)
            #pragma unroll
            for (int nt = 0; nt < 2; ++nt)
                acc[mt][nt] = __builtin_amdgcn_wmma_f32_16x16x32_bf16(
                    false, af[mt], false, bfm[nt], (short)0, acc[mt][nt], false, false);

        // drain this wave's in-flight DMA, then workgroup barrier
        asm volatile("s_wait_asynccnt 0x0" ::: "memory");
        __syncthreads();
        buf ^= 1;
    }

    // epilogue: C layout — VGPR j holds M=j (lanes 0..15) / M=j+8 (lanes 16..31), N=lane%16
    #pragma unroll
    for (int mt = 0; mt < 4; ++mt) {
        #pragma unroll
        for (int nt = 0; nt < 2; ++nt) {
            int rbase = rowBlock + wm * 64 + mt * 16 + hi * 8;
            int cbase = colBlock + wn * 32 + nt * 16 + l16;
            if (Cf) {
                #pragma unroll
                for (int j = 0; j < 8; ++j)
                    Cf[(size_t)(rbase + j) * N + cbase] = acc[mt][nt][j];
            } else {
                #pragma unroll
                for (int j = 0; j < 8; ++j)
                    Cb[(size_t)(rbase + j) * N + cbase] = f2bf(acc[mt][nt][j]);
            }
        }
    }
}

// ---------- K7: mu / logvar head ----------
__global__ __launch_bounds__(256) void head_kernel(
    float* __restrict__ mu, float* __restrict__ logvar,
    const float* __restrict__ h, const float* __restrict__ enc_b, int M) {
    int r = blockIdx.x * blockDim.x + threadIdx.x;
    if (r >= M) return;
    const float* hr = h + (size_t)r * 2 * DCONST;
    float m[DCONST]; float s = 0.f;
    #pragma unroll
    for (int d = 0; d < DCONST; ++d) { m[d] = hr[d] + enc_b[d]; s += m[d] * m[d]; }
    float inv = 1.f / fmaxf(sqrtf(s), 1e-12f);
    #pragma unroll
    for (int d = 0; d < DCONST; ++d) mu[(size_t)r * DCONST + d] = m[d] * inv;
    #pragma unroll
    for (int d = 0; d < DCONST; ++d)
        logvar[(size_t)r * DCONST + d] = -(hr[DCONST + d] + enc_b[DCONST + d]);
}

// ---------- K8: final logits ----------
__global__ __launch_bounds__(256) void logits_kernel(
    float* __restrict__ out, const float* __restrict__ mu,
    const float* __restrict__ items, const float* __restrict__ cates,
    int B, int N) {
    int b = blockIdx.y;
    int n = blockIdx.x * 256 + threadIdx.x;
    __shared__ float mus[KCONST * DCONST];   // 256 floats
    if (threadIdx.x < KCONST * DCONST) {
        int k = threadIdx.x / DCONST, d = threadIdx.x % DCONST;
        mus[threadIdx.x] = mu[((size_t)k * B + b) * DCONST + d];
    }
    __syncthreads();
    if (n >= N) return;
    const float* it = items + (size_t)n * DCONST;
    float p = 0.f;
    #pragma unroll
    for (int k = 0; k < KCONST; ++k) {
        float dot = 0.f;
        #pragma unroll
        for (int d = 0; d < DCONST; ++d) dot += mus[k * DCONST + d] * it[d];
        p += expf(dot) * cates[(size_t)n * KCONST + k];
    }
    out[(size_t)b * N + n] = logf(p);
}

// ---------- launcher ----------
extern "C" void kernel_launch(void* const* d_in, const int* in_sizes, int n_in,
                              void* d_out, int out_size, void* d_ws, size_t ws_size,
                              hipStream_t stream) {
    const float* rating   = (const float*)d_in[0];
    const float* adj      = (const float*)d_in[1];
    const float* item_emb = (const float*)d_in[2];
    const float* proto    = (const float*)d_in[3];
    const float* enc_w    = (const float*)d_in[4];
    const float* enc_b    = (const float*)d_in[5];

    const int twoD = in_sizes[5];          // 128
    const int D    = twoD / 2;             // 64
    const int N    = in_sizes[2] / D;      // 8192
    const int K    = in_sizes[3] / D;      // 4
    const int B    = in_sizes[0] / N;      // 512
    const int M    = K * B;                // 2048
    (void)D; (void)n_in; (void)ws_size; (void)out_size;

    char* ws = (char*)d_ws;
    size_t off = 0;
    auto alloc = [&](size_t bytes) -> void* {
        off = (off + 255) & ~(size_t)255;
        void* p = ws + off;
        off += bytes;
        return p;
    };

    float*  protoN   = (float*) alloc((size_t)K * DCONST * 4);
    float*  items    = (float*) alloc((size_t)N * DCONST * 4);
    float*  cates    = (float*) alloc((size_t)N * K * 4);
    float*  invnorm  = (float*) alloc((size_t)M * 4);
    bf16_t* adjT     = (bf16_t*)alloc((size_t)N * N * 2);
    bf16_t* xk_bf    = (bf16_t*)alloc((size_t)M * N * 2);   // later reused as x_bf
    bf16_t* tmp1_bf  = (bf16_t*)alloc((size_t)M * N * 2);
    float*  tmp2     = (float*) alloc((size_t)M * N * 4);
    bf16_t* encw_bf  = (bf16_t*)alloc((size_t)twoD * N * 2);
    float*  h        = (float*) alloc((size_t)M * twoD * 4);

    float* out_logits = (float*)d_out;
    float* out_mu     = out_logits + (size_t)B * N;
    float* out_logvar = out_mu + (size_t)M * DCONST;

    // prep
    proto_norm_kernel<<<1, 32, 0, stream>>>(protoN, proto);
    items_cates_kernel<<<(N + 255) / 256, 256, 0, stream>>>(items, cates, item_emb, protoN, N);
    transpose_to_bf16<<<dim3(N / 32, N / 32), 256, 0, stream>>>(adjT, adj, N);
    xk_norm_kernel<<<M, 256, 0, stream>>>(invnorm, rating, cates, B, N);
    {
        size_t total = (size_t)M * N;
        xk_build_kernel<<<(unsigned)((total + 255) / 256), 256, 0, stream>>>(xk_bf, rating, cates, invnorm, B, N);
    }
    {
        size_t total = (size_t)twoD * N;
        f2bf_kernel<<<(unsigned)((total + 255) / 256), 256, 0, stream>>>(encw_bf, enc_w, total);
    }

    // two-hop propagation: tmp1 = x_k @ adj (bf16 out), tmp2 = tmp1 @ adj (f32 out)
    dim3 gBig(N / BN, M / BM);
    gemm_bf16_wmma<<<gBig, 256, 0, stream>>>(nullptr, tmp1_bf, xk_bf, adjT, M, N, N);
    gemm_bf16_wmma<<<gBig, 256, 0, stream>>>(tmp2, nullptr, tmp1_bf, adjT, M, N, N);

    // x = (tmp2 + x_k)/2 -> bf16 (reuse xk_bf buffer)
    {
        size_t total = (size_t)M * N;
        combine_kernel<<<(unsigned)((total + 255) / 256), 256, 0, stream>>>(xk_bf, tmp2, rating, cates, invnorm, B, N);
    }

    // encoder: h = x @ enc_w^T   (enc_w is already [2D, N] = BT layout)
    dim3 gEnc(twoD / BN, M / BM);
    gemm_bf16_wmma<<<gEnc, 256, 0, stream>>>(h, nullptr, xk_bf, encw_bf, M, twoD, N);

    // head + final logits
    head_kernel<<<(M + 255) / 256, 256, 0, stream>>>(out_mu, out_logvar, h, enc_b, M);
    logits_kernel<<<dim3(N / 256, B), 256, 0, stream>>>(out_logits, out_mu, items, cates, B, N);
}
